// MultiHeadAttention_44427141710357
// MI455X (gfx1250) — matrix operand order repaired and checked by reading, not compiled, not run on hardware
//
#include <hip/hip_runtime.h>

#define NN 4
#define TT 2048
#define HH 8
#define DD 64
#define DM 512

typedef __attribute__((ext_vector_type(16))) __bf16 v16bf;
typedef __attribute__((ext_vector_type(8)))  __bf16 v8bf;
typedef __attribute__((ext_vector_type(8)))  float  v8f;

// float -> bf16 round-to-nearest-even (bit-level)
static __device__ __forceinline__ __bf16 f2bf(float f) {
  unsigned u = __builtin_bit_cast(unsigned, f);
  u += 0x7fffu + ((u >> 16) & 1u);
  unsigned short h = (unsigned short)(u >> 16);
  return __builtin_bit_cast(__bf16, h);
}

static __device__ __forceinline__ v8f wmma_bf16(v16bf a, v16bf b, v8f c) {
  // 8 args: (neg_a, A, neg_b, B, c_mod, C, reuse_a, reuse_b)
  return __builtin_amdgcn_wmma_f32_16x16x32_bf16(false, a, false, b, (short)0, c, false, false);
}

// A fragment (16x32 bf16): lane holds row (lane&15); K chunks [koff,koff+8) and [koff+16,koff+24), koff=(lane>=16)?8:0
static __device__ __forceinline__ v16bf load_a(const __bf16* base, int stride, int row0, int k0, int lane) {
  const __bf16* p = base + (size_t)(row0 + (lane & 15)) * stride + k0 + ((lane & 16) ? 8 : 0);
  v8bf lo = *(const v8bf*)p;
  v8bf hi = *(const v8bf*)(p + 16);
  v16bf r;
#pragma unroll
  for (int i = 0; i < 8; ++i) { r[i] = lo[i]; r[i + 8] = hi[i]; }
  return r;
}

// B fragment (32x16 bf16): lane holds column (lane&15); K chunks [koff,koff+8) and [koff+16,koff+24), koff=(lane>=16)?8:0
static __device__ __forceinline__ v16bf load_b(const __bf16* base, int stride, int row0, int k0, int lane) {
  const __bf16* p = base + (size_t)(row0 + (lane & 15)) * stride + k0 + ((lane & 16) ? 8 : 0);
  v8bf lo = *(const v8bf*)p;
  v8bf hi = *(const v8bf*)(p + 16);
  v16bf r;
#pragma unroll
  for (int i = 0; i < 8; ++i) { r[i] = lo[i]; r[i + 8] = hi[i]; }
  return r;
}

__global__ void cvt_f32_bf16(const float* __restrict__ in, __bf16* __restrict__ outp, int nelem) {
  int i = blockIdx.x * 256 + threadIdx.x;
  if (i < nelem) outp[i] = f2bf(in[i]);
}

// QKV projection + RoPE, split by matrix: each wave computes ONE of q/k/v for a 16(t) x 64(d) strip.
// 4 accumulators per wave -> double-buffered fragments fit in registers (real load/WMMA overlap).
// All 8 waves in a block share (mat, head) -> identical weight fragments per K-step (WGP$/L2 reuse).
__global__ __launch_bounds__(256) void qkv_rope_kernel(
    const __bf16* __restrict__ xb, const __bf16* __restrict__ wb,
    __bf16* __restrict__ qo, __bf16* __restrict__ ko, __bf16* __restrict__ vto) {
  const int lane = threadIdx.x & 31;
  const int wid  = threadIdx.x >> 5;
  const int mh   = blockIdx.x >> 6;                 // 0..23 = mat*8 + h  (64 blocks each)
  const int mat  = mh >> 3;                         // 0=q 1=k 2=v
  const int h    = mh & 7;
  const int sb   = ((blockIdx.x & 63) << 3) + wid;  // 0..511 strips over N*(T/16)
  const int n    = sb >> 7;
  const int t0   = (sb & 127) << 4;
  const int wrow0 = mat * DM + h * DD;

  const v8f vz = {0.f, 0.f, 0.f, 0.f, 0.f, 0.f, 0.f, 0.f};
  v8f acc[4] = {vz, vz, vz, vz};

  const __bf16* xrow = xb + (size_t)(n * TT + t0) * DM;

  // software pipeline: fragments for step k in (ac, bc); loads for k+32 issued before the WMMAs retire
  v16bf ac = load_a(xrow, DM, 0, 0, lane);
  v16bf bc[4];
#pragma unroll
  for (int j = 0; j < 4; ++j) bc[j] = load_b(wb, DM, wrow0 + 16 * j, 0, lane);

#pragma unroll 2
  for (int k0 = 0; k0 < DM; k0 += 32) {
    const int kn = (k0 + 32 < DM) ? k0 + 32 : 0;    // wrapped dummy load on last step (in-bounds, discarded)
    v16bf an = load_a(xrow, DM, 0, kn, lane);
    v16bf bn[4];
#pragma unroll
    for (int j = 0; j < 4; ++j) bn[j] = load_b(wb, DM, wrow0 + 16 * j, kn, lane);
#pragma unroll
    for (int j = 0; j < 4; ++j) acc[j] = wmma_bf16(ac, bc[j], acc[j]);
    ac = an;
#pragma unroll
    for (int j = 0; j < 4; ++j) bc[j] = bn[j];
  }

  const int dl  = lane & 15;
  const int hi8 = (lane & 16) ? 8 : 0;
  const size_t nhh = (size_t)(n * HH + h);

  if (mat == 2) {
    // V: store transposed [n,h,d,t]
    __bf16* vbase = vto + nhh * DD * TT;
#pragma unroll
    for (int j = 0; j < 4; ++j) {
      const int d = 16 * j + dl;
#pragma unroll
      for (int r = 0; r < 8; ++r)
        vbase[(size_t)d * TT + (t0 + r + hi8)] = f2bf(acc[j][r]);
    }
  } else {
    // Q or K: apply RoPE then store [n,h,t,d]
    const float kln  = 9.210340371976184f / 32.f;       // ln(10000)/32
    const float inv0 = __expf(-(float)dl * kln);        // 10000^(-dl/32)
    const float inv1 = __expf(-(float)(dl + 16) * kln); // 10000^(-(dl+16)/32)
#pragma unroll
    for (int r = 0; r < 8; ++r) {
      const float tp = (float)(t0 + r + hi8);
      const float th0 = tp * inv0, th1 = tp * inv1;
      const float c0 = __cosf(th0), s0 = __sinf(th0);
      const float c1 = __cosf(th1), s1 = __sinf(th1);
      float u0 = acc[0][r], u1 = acc[1][r], u2 = acc[2][r], u3 = acc[3][r];
      acc[0][r] = u0 * c0 - u2 * s0;
      acc[1][r] = u1 * c1 - u3 * s1;
      acc[2][r] = u2 * c0 + u0 * s0;
      acc[3][r] = u3 * c1 + u1 * s1;
    }
    __bf16* obase = (mat == 0 ? qo : ko) + nhh * TT * DD;
#pragma unroll
    for (int j = 0; j < 4; ++j) {
      const int d = 16 * j + dl;
#pragma unroll
      for (int r = 0; r < 8; ++r)
        obase[(size_t)(t0 + r + hi8) * DD + d] = f2bf(acc[j][r]);
    }
  }
}

// Banded attention, flash-style. One wave = 16 queries x full head dim (64).
// Band diff = tk - tq in [-127, 128]; key loop clipped to [t0-127, t0+143] rounded to 32.
// K-tile fragments are double-buffered across chunks; V loads overlap the softmax + LDS transpose.
__global__ __launch_bounds__(256) void attn_kernel(
    const __bf16* __restrict__ qb, const __bf16* __restrict__ kb,
    const __bf16* __restrict__ vt, __bf16* __restrict__ ao) {
  __shared__ __bf16 plds[8 * 16 * 32];   // 1KB P tile per wave
  const int lane = threadIdx.x & 31;
  const int wid  = threadIdx.x >> 5;
  const int gw   = blockIdx.x * 8 + wid;
  const int nh   = gw >> 7;              // (n*H + h)
  const int t0   = (gw & 127) << 4;
  const int n    = nh >> 3;
  const int h    = nh & 7;
  const int dl   = lane & 15;
  const int hi8  = (lane & 16) ? 8 : 0;

  const __bf16* qrow = qb + ((size_t)nh * TT + t0) * DD;
  const v16bf qf0 = load_a(qrow, DD, 0, 0, lane);
  const v16bf qf1 = load_a(qrow, DD, 0, 32, lane);

  const v8f vz = {0.f, 0.f, 0.f, 0.f, 0.f, 0.f, 0.f, 0.f};
  v8f out[4] = {vz, vz, vz, vz};
  float m[8], l[8];
#pragma unroll
  for (int r = 0; r < 8; ++r) { m[r] = -3.0e38f; l[r] = 0.f; }

  int ks = t0 - 127; if (ks < 0) ks = 0; ks &= ~31;
  int ke = t0 + 144; if (ke > TT) ke = TT; ke = (ke + 31) & ~31;

  const __bf16* krows = kb + (size_t)nh * TT * DD;
  const __bf16* vrows = vt + (size_t)nh * DD * TT;
  __bf16* pl = plds + wid * 512;
  const float scale = 0.125f;            // 1/sqrt(64)

  // preload first K chunk fragments
  v16bf kf0 = load_b(krows, DD, ks,      0,  lane);
  v16bf kf1 = load_b(krows, DD, ks,      32, lane);
  v16bf kf2 = load_b(krows, DD, ks + 16, 0,  lane);
  v16bf kf3 = load_b(krows, DD, ks + 16, 32, lane);

#pragma unroll 1
  for (int kc = ks; kc < ke; kc += 32) {
    v8f s0 = vz, s1 = vz;
    s0 = wmma_bf16(qf0, kf0, s0);
    s0 = wmma_bf16(qf1, kf1, s0);
    s1 = wmma_bf16(qf0, kf2, s1);
    s1 = wmma_bf16(qf1, kf3, s1);

    // V^T fragments for this chunk: independent of softmax, loads fly during VALU + LDS work
    v16bf vf[4];
#pragma unroll
    for (int j = 0; j < 4; ++j) vf[j] = load_b(vrows, TT, 16 * j, kc, lane);

    // prefetch next chunk's K fragments (wrapped dummy on last iteration)
    const int kn = (kc + 32 < ke) ? kc + 32 : ks;
    kf0 = load_b(krows, DD, kn,      0,  lane);
    kf1 = load_b(krows, DD, kn,      32, lane);
    kf2 = load_b(krows, DD, kn + 16, 0,  lane);
    kf3 = load_b(krows, DD, kn + 16, 32, lane);

    const int tk0 = kc + dl;
    const int tk1 = tk0 + 16;
    float p0[8], p1[8];
#pragma unroll
    for (int r = 0; r < 8; ++r) {
      const int tq = t0 + r + hi8;
      const float sv0 = s0[r] * scale;
      const float sv1 = s1[r] * scale;
      const int d0 = tk0 - tq;
      const int d1 = tk1 - tq;
      const bool ok0 = (d0 >= -127) && (d0 <= 128) && (tk0 < TT);
      const bool ok1 = (d1 >= -127) && (d1 <= 128) && (tk1 < TT);
      float a0 = ok0 ? sv0 : -3.0e38f;
      float a1 = ok1 ? sv1 : -3.0e38f;
      float mx = fmaxf(a0, a1);
      mx = fmaxf(mx, __shfl_xor(mx, 1, 32));
      mx = fmaxf(mx, __shfl_xor(mx, 2, 32));
      mx = fmaxf(mx, __shfl_xor(mx, 4, 32));
      mx = fmaxf(mx, __shfl_xor(mx, 8, 32));
      const float mn = fmaxf(m[r], mx);
      const float e0 = ok0 ? __expf(sv0 - mn) : 0.f;
      const float e1 = ok1 ? __expf(sv1 - mn) : 0.f;
      float rs = e0 + e1;
      rs += __shfl_xor(rs, 1, 32);
      rs += __shfl_xor(rs, 2, 32);
      rs += __shfl_xor(rs, 4, 32);
      rs += __shfl_xor(rs, 8, 32);
      const float alpha = __expf(m[r] - mn);
      l[r] = l[r] * alpha + rs;
      m[r] = mn;
      p0[r] = e0; p1[r] = e1;
#pragma unroll
      for (int j = 0; j < 4; ++j) out[j][r] = out[j][r] * alpha;
    }

    // D-layout -> A-layout transpose of P through LDS (private 1KB tile per wave)
#pragma unroll
    for (int r = 0; r < 8; ++r) {
      pl[(r + hi8) * 32 + dl]      = f2bf(p0[r]);
      pl[(r + hi8) * 32 + dl + 16] = f2bf(p1[r]);
    }
    asm volatile("s_wait_dscnt 0" ::: "memory");   // CDNA5 split-counter LDS wait (same-wave RAW)
    v16bf pf = load_a((const __bf16*)pl, 32, 0, 0, lane);

#pragma unroll
    for (int j = 0; j < 4; ++j) out[j] = wmma_bf16(pf, vf[j], out[j]);
  }

  float linv[8];
#pragma unroll
  for (int r = 0; r < 8; ++r) linv[r] = 1.0f / l[r];   // diag key always valid => l > 0
#pragma unroll
  for (int j = 0; j < 4; ++j) {
    const int d = h * DD + 16 * j + dl;
#pragma unroll
    for (int r = 0; r < 8; ++r) {
      const int t = t0 + r + hi8;
      ao[((size_t)n * TT + t) * DM + d] = f2bf(out[j][r] * linv[r]);
    }
  }
}

// Output projection: out[8192,512] = attn_bf16 @ out_w^T + out_b  (fp32 out)
// 8 waves of a block cover the 8 column groups of the same row strip (shared A rows);
// K-loop fragments double-buffered.
__global__ __launch_bounds__(256) void outproj_kernel(
    const __bf16* __restrict__ ab, const __bf16* __restrict__ wb,
    const float* __restrict__ bias, float* __restrict__ out) {
  const int lane = threadIdx.x & 31;
  const int wid  = threadIdx.x >> 5;
  const int gw   = blockIdx.x * 8 + wid;
  const int cg   = gw & 7;               // 8 column groups of 64
  const int row0 = (gw >> 3) << 4;       // 512 row strips of 16
  const int wrow0 = cg * DD;
  const v8f vz = {0.f, 0.f, 0.f, 0.f, 0.f, 0.f, 0.f, 0.f};
  v8f acc[4] = {vz, vz, vz, vz};

  v16bf ac = load_a(ab, DM, row0, 0, lane);
  v16bf bc[4];
#pragma unroll
  for (int j = 0; j < 4; ++j) bc[j] = load_b(wb, DM, wrow0 + 16 * j, 0, lane);

#pragma unroll 2
  for (int k0 = 0; k0 < DM; k0 += 32) {
    const int kn = (k0 + 32 < DM) ? k0 + 32 : 0;
    v16bf an = load_a(ab, DM, row0, kn, lane);
    v16bf bn[4];
#pragma unroll
    for (int j = 0; j < 4; ++j) bn[j] = load_b(wb, DM, wrow0 + 16 * j, kn, lane);
#pragma unroll
    for (int j = 0; j < 4; ++j) acc[j] = wmma_bf16(ac, bc[j], acc[j]);
    ac = an;
#pragma unroll
    for (int j = 0; j < 4; ++j) bc[j] = bn[j];
  }

  const int dl  = lane & 15;
  const int hi8 = (lane & 16) ? 8 : 0;
#pragma unroll
  for (int j = 0; j < 4; ++j) {
    const int c = wrow0 + 16 * j + dl;
    const float b = bias[c];
#pragma unroll
    for (int r = 0; r < 8; ++r) {
      out[(size_t)(row0 + r + hi8) * DM + c] = acc[j][r] + b;
    }
  }
}

extern "C" void kernel_launch(void* const* d_in, const int* in_sizes, int n_in,
                              void* d_out, int out_size, void* d_ws, size_t ws_size,
                              hipStream_t stream) {
  (void)n_in; (void)out_size; (void)ws_size;
  const float* x    = (const float*)d_in[0];
  const float* wqkv = (const float*)d_in[1];
  const float* ow   = (const float*)d_in[2];
  const float* ob   = (const float*)d_in[3];
  float* out = (float*)d_out;

  char* ws = (char*)d_ws;
  __bf16* xb  = (__bf16*)ws; ws += (size_t)NN * TT * DM * 2;       // x in bf16
  __bf16* wqb = (__bf16*)ws; ws += (size_t)3 * DM * DM * 2;        // Wqkv bf16
  __bf16* owb = (__bf16*)ws; ws += (size_t)DM * DM * 2;            // out_w bf16
  __bf16* qb  = (__bf16*)ws; ws += (size_t)NN * HH * TT * DD * 2;  // Q (roped)   [n,h,t,d]
  __bf16* kb  = (__bf16*)ws; ws += (size_t)NN * HH * TT * DD * 2;  // K (roped)   [n,h,t,d]
  __bf16* vt  = (__bf16*)ws; ws += (size_t)NN * HH * TT * DD * 2;  // V transposed [n,h,d,t]
  __bf16* ab  = (__bf16*)ws; ws += (size_t)NN * TT * DM * 2;       // attn output [n,t,h*d]

  const int nx = in_sizes[0];  // 4*2048*512
  const int nw = in_sizes[1];  // 1536*512
  const int no = in_sizes[2];  // 512*512

  cvt_f32_bf16<<<(nx + 255) / 256, 256, 0, stream>>>(x, xb, nx);
  cvt_f32_bf16<<<(nw + 255) / 256, 256, 0, stream>>>(wqkv, wqb, nw);
  cvt_f32_bf16<<<(no + 255) / 256, 256, 0, stream>>>(ow, owb, no);

  // QKV: 3 mats x 8 heads x 512 strips, 8 waves/block sharing (mat, head)
  qkv_rope_kernel<<<3 * HH * (NN * (TT / 16) / 8), 256, 0, stream>>>(xb, wqb, qb, kb, vt);
  // attention: 4096 waves, 8 waves/block over consecutive t-strips of one (n,h)
  attn_kernel<<<(NN * HH * (TT / 16)) / 8, 256, 0, stream>>>(qb, kb, vt, ab);
  // projection: 4096 waves
  outproj_kernel<<<(NN * (TT / 16) * 8) / 8, 256, 0, stream>>>(ab, owb, ob, out);

  (void)ob;
}
